// LearnableWaveletPacket_47708496724243
// MI455X (gfx1250) — matrix operand order repaired and checked by reading, not compiled, not run on hardware
//
#include <hip/hip_runtime.h>
#include <hip/hip_bf16.h>

typedef __attribute__((ext_vector_type(16))) _Float16 v16h;
typedef __attribute__((ext_vector_type(8)))  float    v8f;

// ---------------------------------------------------------------------------
// Fused one-level wavelet packet transform, one wave per 16x16 output tile.
//
// Row conv :  T = In(16x32) x Kb(32x16)        (Kb built from dec_lo/dec_hi)
// Col conv :  S = Kt(16x32) x T(32x16)
// Both stages are V_WMMA_F32_16X16X32_F16; the 32x16 intermediate T is staged
// through LDS (fp16) to convert accumulator layout -> B-matrix layout.
// All 4 subbands (ll,lh,hl,hh) of the tile are produced by 8 WMMAs total.
// ---------------------------------------------------------------------------
__global__ __launch_bounds__(32)
void wavelet_level_kernel(const float* in_base,     // level0: x, else: d_out
                          float*       out,         // d_out (features region)
                          const float* dec_lo,
                          const float* dec_hi,
                          int Hin, int Hout, int level,
                          int inOff0, int inOff1, int inOff2,
                          int outOff0, int outOff1, int outOff2)
{
    const int bc = blockIdx.z;
    const int b  = bc / 3;
    const int c  = bc % 3;

    const float* in;
    if (level == 0) {
        in = in_base + (size_t)bc * Hin * Hin;
    } else {
        const int io = (c == 0) ? inOff0 : (c == 1) ? inOff1 : inOff2;
        // LL of previous level == subband 0 of the previous feature tensor
        in = in_base + io + (size_t)(b * 4) * Hin * Hin;
    }
    const int oo = (c == 0) ? outOff0 : (c == 1) ? outOff1 : outOff2;
    float* outT  = out + oo + (size_t)(b * 4) * Hout * Hout;

    const int x0   = blockIdx.x * 16;
    const int y0   = blockIdx.y * 16;
    const int lane = threadIdx.x;
    const int l15  = lane & 15;
    const bool hiHalf = lane >= 16;

    const float k0l = dec_lo[0], k1l = dec_lo[1];
    const float k0h = dec_hi[0], k1h = dec_hi[1];

    // conv (2-tap, stride 2, pad 1):  out[j] = k0*in[2j-1] + k1*in[2j]
    const int inRow0 = 2 * y0 - 1;
    const int inCol0 = 2 * x0 - 1;

    // ---- Stage 1: A fragments = input tile rows (16x32 fp16, WMMA A layout)
    // lane l: M = l&15 ; K = h + (h<8 ? (hi?8:0) : (hi?16:8))
    v16h a_top, a_bot;
#pragma unroll
    for (int h = 0; h < 16; ++h) {
        const int K    = h + (h < 8 ? (hiHalf ? 8 : 0) : (hiHalf ? 16 : 8));
        const int colg = inCol0 + K;
        const int rt   = inRow0 + l15;
        const int rb   = rt + 16;
        const bool cok = (colg >= 0) && (colg < Hin);
        const float vt = (cok && rt >= 0 && rt < Hin) ? in[(size_t)rt * Hin + colg] : 0.0f;
        const float vb = (cok && rb >= 0 && rb < Hin) ? in[(size_t)rb * Hin + colg] : 0.0f;
        a_top[h] = (_Float16)vt;
        a_bot[h] = (_Float16)vb;
    }

    // ---- Stage 1: B fragments = banded row-filter matrices Kb(32x16)
    // B layout: lane n = l&15 ; K = h + (h<8 ? (hi?8:0) : (hi?16:8)) ; Kb[2n,n]=k0, Kb[2n+1,n]=k1
    v16h b_lo, b_hi;
#pragma unroll
    for (int h = 0; h < 16; ++h) {
        const int K = h + (h < 8 ? (hiHalf ? 8 : 0) : (hiHalf ? 16 : 8));
        const int n = l15;
        b_lo[h] = (_Float16)((K == 2 * n) ? k0l : (K == 2 * n + 1) ? k1l : 0.0f);
        b_hi[h] = (_Float16)((K == 2 * n) ? k0h : (K == 2 * n + 1) ? k1h : 0.0f);
    }

    const v8f z = {};
    v8f tlo_t = __builtin_amdgcn_wmma_f32_16x16x32_f16(false, a_top, false, b_lo, (short)0, z, false, false);
    v8f tlo_b = __builtin_amdgcn_wmma_f32_16x16x32_f16(false, a_bot, false, b_lo, (short)0, z, false, false);
    v8f thi_t = __builtin_amdgcn_wmma_f32_16x16x32_f16(false, a_top, false, b_hi, (short)0, z, false, false);
    v8f thi_b = __builtin_amdgcn_wmma_f32_16x16x32_f16(false, a_bot, false, b_hi, (short)0, z, false, false);

    // ---- Relayout 32x16 intermediates via LDS (accumulator -> B layout)
    __shared__ _Float16 lds_lo[32 * 16];
    __shared__ _Float16 lds_hi[32 * 16];
#pragma unroll
    for (int r = 0; r < 8; ++r) {
        const int row = r + (hiHalf ? 8 : 0);   // accumulator: M = r (+8 for hi lanes)
        const int col = l15;
        lds_lo[row * 16 + col]        = (_Float16)tlo_t[r];
        lds_lo[(row + 16) * 16 + col] = (_Float16)tlo_b[r];
        lds_hi[row * 16 + col]        = (_Float16)thi_t[r];
        lds_hi[(row + 16) * 16 + col] = (_Float16)thi_b[r];
    }
    __syncthreads();

    // ---- Stage 2: A fragments = banded column-filter matrices Kt(16x32)
    v16h ac_lo, ac_hi;
#pragma unroll
    for (int h = 0; h < 16; ++h) {
        const int K = h + (h < 8 ? (hiHalf ? 8 : 0) : (hiHalf ? 16 : 8));
        const int m = l15;
        ac_lo[h] = (_Float16)((K == 2 * m) ? k0l : (K == 2 * m + 1) ? k1l : 0.0f);
        ac_hi[h] = (_Float16)((K == 2 * m) ? k0h : (K == 2 * m + 1) ? k1h : 0.0f);
    }

    // ---- Stage 2: B fragments = T tiles gathered from LDS
    v16h bT_lo, bT_hi;
#pragma unroll
    for (int h = 0; h < 16; ++h) {
        const int K = h + (h < 8 ? (hiHalf ? 8 : 0) : (hiHalf ? 16 : 8));
        bT_lo[h] = lds_lo[K * 16 + l15];
        bT_hi[h] = lds_hi[K * 16 + l15];
    }

    v8f ll = __builtin_amdgcn_wmma_f32_16x16x32_f16(false, ac_lo, false, bT_lo, (short)0, z, false, false);
    v8f lh = __builtin_amdgcn_wmma_f32_16x16x32_f16(false, ac_hi, false, bT_lo, (short)0, z, false, false);
    v8f hl = __builtin_amdgcn_wmma_f32_16x16x32_f16(false, ac_lo, false, bT_hi, (short)0, z, false, false);
    v8f hh = __builtin_amdgcn_wmma_f32_16x16x32_f16(false, ac_hi, false, bT_hi, (short)0, z, false, false);

    // ---- Store 4 subbands; accumulator element r = (M = r + (hi?8:0), N = l15)
    const size_t plane = (size_t)Hout * Hout;
#pragma unroll
    for (int r = 0; r < 8; ++r) {
        const int row = y0 + r + (hiHalf ? 8 : 0);
        const int col = x0 + l15;
        if (row < Hout && col < Hout) {
            const size_t p = (size_t)row * Hout + col;
            outT[0 * plane + p] = ll[r];
            outT[1 * plane + p] = lh[r];
            outT[2 * plane + p] = hl[r];
            outT[3 * plane + p] = hh[r];
        }
    }
}

// ---------------------------------------------------------------------------
// Antialiased linear resize of the 9 feature tensors [16,4,S,S] -> [16,4,224,224].
// Triangle kernel scaled by min(1, 224/S); per-axis edge renormalization
// (matches jax.image.resize(..., "linear", antialias=True)).
// Sources are small and hot in L2 (192 MB), so gather loads are cheap.
// ---------------------------------------------------------------------------
struct Off9 { int o[9]; };

__global__ __launch_bounds__(256)
void resize_bilinear_aa_kernel(const float* feats, float* outR, Off9 off, int total)
{
    const int idx = blockIdx.x * blockDim.x + threadIdx.x;
    if (idx >= total) return;

    const int OUT = 224;
    const int per = 16 * 4 * OUT * OUT;
    const int t   = idx / per;
    const int r   = idx - t * per;
    const int ox  = r % OUT;
    const int oy  = (r / OUT) % OUT;
    const int bs  = r / (OUT * OUT);            // b*4 + subband

    const int lvl = t % 3;
    const int S   = (lvl == 0) ? 257 : (lvl == 1) ? 129 : 65;
    const float* src = feats + off.o[t] + (size_t)bs * S * S;

    const float scale = 224.0f / (float)S;
    const float ks    = scale < 1.0f ? scale : 1.0f;   // antialias kernel scale
    const float rad   = 1.0f / ks;

    // x-axis weights (<= 4 taps)
    const float cx = (ox + 0.5f) / scale - 0.5f;
    int xs = (int)ceilf(cx - rad);
    int xe = (int)floorf(cx + rad);
    if (xs < 0) xs = 0;
    if (xe > S - 1) xe = S - 1;
    int nx = xe - xs + 1; if (nx > 4) nx = 4;
    float wx[4]; float sumx = 0.0f;
#pragma unroll
    for (int i = 0; i < 4; ++i) {
        float w = 0.0f;
        if (i < nx) {
            const float d = ((float)(xs + i) - cx) * ks;
            w = 1.0f - fabsf(d);
            if (w < 0.0f) w = 0.0f;
        }
        wx[i] = w; sumx += w;
    }

    // y-axis weights (<= 4 taps)
    const float cy = (oy + 0.5f) / scale - 0.5f;
    int ys = (int)ceilf(cy - rad);
    int ye = (int)floorf(cy + rad);
    if (ys < 0) ys = 0;
    if (ye > S - 1) ye = S - 1;
    int ny = ye - ys + 1; if (ny > 4) ny = 4;
    float wy[4]; float sumy = 0.0f;
#pragma unroll
    for (int j = 0; j < 4; ++j) {
        float w = 0.0f;
        if (j < ny) {
            const float d = ((float)(ys + j) - cy) * ks;
            w = 1.0f - fabsf(d);
            if (w < 0.0f) w = 0.0f;
        }
        wy[j] = w; sumy += w;
    }

    float acc = 0.0f;
#pragma unroll
    for (int j = 0; j < 4; ++j) {
        if (wy[j] != 0.0f) {
            const float* row = src + (size_t)(ys + j) * S;
            float ax = 0.0f;
#pragma unroll
            for (int i = 0; i < 4; ++i) {
                if (wx[i] != 0.0f) ax += wx[i] * row[xs + i];
            }
            acc += wy[j] * ax;
        }
    }
    outR[idx] = acc / (sumx * sumy);
}

// ---------------------------------------------------------------------------
extern "C" void kernel_launch(void* const* d_in, const int* in_sizes, int n_in,
                              void* d_out, int out_size, void* d_ws, size_t ws_size,
                              hipStream_t stream)
{
    (void)in_sizes; (void)n_in; (void)out_size; (void)d_ws; (void)ws_size;

    const float* x   = (const float*)d_in[0];
    const float* dlo = (const float*)d_in[1];
    const float* dhi = (const float*)d_in[2];
    float*       out = (float*)d_out;

    const int H0 = 512, H1 = 257, H2 = 129, H3 = 65;
    const int S0 = 64 * H1 * H1;   // elems per [16,4,257,257] tensor
    const int S1 = 64 * H2 * H2;
    const int S2 = 64 * H3 * H3;

    // feature tensor offsets: order t = c*3 + lvl
    int off[9]; int acc = 0;
    for (int c = 0; c < 3; ++c) {
        off[c * 3 + 0] = acc; acc += S0;
        off[c * 3 + 1] = acc; acc += S1;
        off[c * 3 + 2] = acc; acc += S2;
    }
    const int resizedBase = acc;   // 16,687,680 floats

    // Level 0: x(512) -> subbands(257)
    {
        dim3 g((H1 + 15) / 16, (H1 + 15) / 16, 48);
        hipLaunchKernelGGL(wavelet_level_kernel, g, dim3(32), 0, stream,
                           x, out, dlo, dhi, H0, H1, 0,
                           0, 0, 0, off[0], off[3], off[6]);
    }
    // Level 1: LL1(257) -> subbands(129)
    {
        dim3 g((H2 + 15) / 16, (H2 + 15) / 16, 48);
        hipLaunchKernelGGL(wavelet_level_kernel, g, dim3(32), 0, stream,
                           out, out, dlo, dhi, H1, H2, 1,
                           off[0], off[3], off[6], off[1], off[4], off[7]);
    }
    // Level 2: LL2(129) -> subbands(65)
    {
        dim3 g((H3 + 15) / 16, (H3 + 15) / 16, 48);
        hipLaunchKernelGGL(wavelet_level_kernel, g, dim3(32), 0, stream,
                           out, out, dlo, dhi, H2, H3, 2,
                           off[1], off[4], off[7], off[2], off[5], off[8]);
    }
    // Resize all 9 tensors to 224x224
    {
        const int total = 9 * 16 * 4 * 224 * 224;
        Off9 o; for (int i = 0; i < 9; ++i) o.o[i] = off[i];
        hipLaunchKernelGGL(resize_bilinear_aa_kernel,
                           dim3((total + 255) / 256), dim3(256), 0, stream,
                           out, out + resizedBase, o, total);
    }
}
